// ScaledDotProAttention_5557687681458
// MI455X (gfx1250) — matrix operand order repaired and checked by reading, not compiled, not run on hardware
//
#include <hip/hip_runtime.h>

// Scaled dot-product attention for B=16, S=2048, D=128 (fp32 in/out).
// d_out = context [16,2048,128] followed by attention [16,2048,2048], flat f32.
//
// Strategy: one workgroup per (batch, 16-row query tile). Scores/softmax in
// f32 LDS; matmuls via v_wmma_f32_16x16x32_f16 (f16 A/B, f32 accumulate).

#define BB 16
#define SS 2048
#define DD 128
#define NTHREADS 128

static constexpr float SM_SCALE = 0.08838834764831845f; // 1/sqrt(128)

typedef _Float16 h16 __attribute__((ext_vector_type(16)));
typedef _Float16 h8  __attribute__((ext_vector_type(8)));
typedef float    f8  __attribute__((ext_vector_type(8)));
typedef float    f4  __attribute__((ext_vector_type(4)));

// LDS layout (bytes):
//   sc   : f32 [16][2048]            131072
//   pf   : f16 [16][2048]             65536
//   qh   : f16 [16][128]               4096
//   red  : f32 [16*8 + 16]              576
#define LDS_SC_OFF   0
#define LDS_PF_OFF   (16 * SS * 4)
#define LDS_QH_OFF   (LDS_PF_OFF + 16 * SS * 2)
#define LDS_RED_OFF  (LDS_QH_OFF + 16 * DD * 2)
#define LDS_BYTES    (LDS_RED_OFF + (16 * 8 + 16) * 4)

__global__ __launch_bounds__(NTHREADS, 1)
void attn_fwd_kernel(const float* __restrict__ Q,
                     const float* __restrict__ K,
                     const float* __restrict__ V,
                     float* __restrict__ ctx_out,
                     float* __restrict__ att_out)
{
    extern __shared__ char smem[];
    float*    sc  = (float*)(smem + LDS_SC_OFF);    // raw scores -> exp values
    _Float16* pf  = (_Float16*)(smem + LDS_PF_OFF); // normalized P, f16
    _Float16* qh  = (_Float16*)(smem + LDS_QH_OFF); // Q tile, f16
    float*    red = (float*)(smem + LDS_RED_OFF);   // reductions + inv sums

    const int tid  = threadIdx.x;
    const int wave = tid >> 5;
    const int lane = tid & 31;

    const int wg = blockIdx.x;
    const int b  = wg >> 7;            // 128 query tiles per batch
    const int q0 = (wg & 127) << 4;    // first query row of this tile

    // ---- Stage Q tile (16 x 128) into LDS as f16 (coalesced) ----
    const float* Qt = Q + ((size_t)b * SS + q0) * DD;
    for (int i = tid; i < 16 * DD; i += NTHREADS)
        qh[i] = (_Float16)Qt[i];
    __syncthreads();

    const int r  = lane & 15;   // row (A/C) or column (B) index
    const int hi = lane >> 4;   // half-wave select

    // ---- Build the 4 Q A-fragments (K-chunks of 32) once per wave ----
    // A 16x32 here in contiguous k order (B matches): lanes 0-15 hold K = k0+[0..15];
    // lanes 16-31 hold K = k0+[16..31].
    h16 aq[4];
#pragma unroll
    for (int c = 0; c < 4; ++c) {
        const _Float16* p = qh + r * DD + c * 32 + hi * 16;
        h8 lo = *(const h8*)(p);
        h8 up = *(const h8*)(p + 8);
#pragma unroll
        for (int j = 0; j < 8; ++j) { aq[c][j] = lo[j]; aq[c][8 + j] = up[j]; }
    }

    // ---- Pass 1: scores = Q @ K^T (raw, unscaled) into LDS ----
    const float* Kb = K + (size_t)b * SS * DD;
    for (int kt = wave; kt < SS / 16; kt += 4) {
        f8 acc = {};
#pragma unroll
        for (int c = 0; c < 4; ++c) {
            // B fragment (32x16): column n = key row kt*16+n; row j = feature.
            // lane n (+16): features c*32 + hi*16 + [0..15] of key kt*16+n.
            const float* kp = Kb + (size_t)(kt * 16 + r) * DD + c * 32 + hi * 16;
            f4 x0 = *(const f4*)(kp + 0);
            f4 x1 = *(const f4*)(kp + 4);
            f4 x2 = *(const f4*)(kp + 8);
            f4 x3 = *(const f4*)(kp + 12);
            h16 bk;
#pragma unroll
            for (int j = 0; j < 4; ++j) {
                bk[j]      = (_Float16)x0[j];
                bk[4 + j]  = (_Float16)x1[j];
                bk[8 + j]  = (_Float16)x2[j];
                bk[12 + j] = (_Float16)x3[j];
            }
            acc = __builtin_amdgcn_wmma_f32_16x16x32_f16(
                false, aq[c], false, bk, (short)0, acc, false, false);
        }
        // C layout: element m -> row m + 8*hi, col = kt*16 + (lane&15)
#pragma unroll
        for (int m = 0; m < 8; ++m)
            sc[(m + hi * 8) * SS + kt * 16 + r] = acc[m];
    }
    __syncthreads();

    // ---- Pass 2: row softmax over S=2048 (8 threads per row) ----
    const int srow = tid >> 3;
    const int sseg = tid & 7;
    float* rowp = sc + srow * SS + sseg * 256;

    float m = -3.402823466e38f;
    for (int i = 0; i < 256; ++i) m = fmaxf(m, rowp[i]);
    red[srow * 8 + sseg] = m;
    __syncthreads();

    float rowmax = red[srow * 8];
#pragma unroll
    for (int i = 1; i < 8; ++i) rowmax = fmaxf(rowmax, red[srow * 8 + i]);
    __syncthreads();  // done reading maxes before red is reused for sums

    float s = 0.f;
    for (int i = 0; i < 256; ++i) {
        float e = __expf((rowp[i] - rowmax) * SM_SCALE);
        rowp[i] = e;
        s += e;
    }
    red[srow * 8 + sseg] = s;
    __syncthreads();

    if (tid < 16) {
        float t = 0.f;
#pragma unroll
        for (int i = 0; i < 8; ++i) t += red[tid * 8 + i];
        red[128 + tid] = 1.f / t;
    }
    __syncthreads();

    // ---- Normalize; write attention (coalesced); stash P as f16 ----
    float* attb = att_out + ((size_t)b * SS + q0) * SS;
    for (int idx = tid; idx < 16 * SS; idx += NTHREADS) {
        const int row = idx >> 11;             // idx / 2048
        float p = sc[idx] * red[128 + row];
        attb[idx] = p;
        pf[idx]   = (_Float16)p;
    }
    __syncthreads();

    // ---- Pass 3: context = P @ V; each wave owns 2 of 8 N-tiles ----
    const float* Vb = V + (size_t)b * SS * DD;
    const int n0a = wave * 16;
    const int n0b = (wave + 4) * 16;
    f8 acc0 = {}, acc1 = {};

    for (int c = 0; c < SS / 32; ++c) {
        // A fragment of P (16 x 32 chunk) from LDS
        const _Float16* pp = pf + r * SS + c * 32 + hi * 16;
        h8 lo = *(const h8*)(pp);
        h8 up = *(const h8*)(pp + 8);
        h16 ap;
#pragma unroll
        for (int j = 0; j < 8; ++j) { ap[j] = lo[j]; ap[8 + j] = up[j]; }

        // B fragments of V (32 x 16): row j = key c*32 + hi*16 + j,
        // column n = feature n0 + (lane&15). Lanes 0-15 read 16 consecutive
        // floats per row -> coalesced 64B segments.
        const float* vp = Vb + (size_t)(c * 32 + hi * 16) * DD + r;
        h16 bv0, bv1;
#pragma unroll
        for (int j = 0; j < 16; ++j) {
            const float* vr = vp + (size_t)j * DD;
            bv0[j] = (_Float16)vr[n0a];
            bv1[j] = (_Float16)vr[n0b];
        }
        acc0 = __builtin_amdgcn_wmma_f32_16x16x32_f16(
            false, ap, false, bv0, (short)0, acc0, false, false);
        acc1 = __builtin_amdgcn_wmma_f32_16x16x32_f16(
            false, ap, false, bv1, (short)0, acc1, false, false);
    }

    // ---- Store context (f32) ----
    float* ctxb = ctx_out + ((size_t)b * SS + q0) * DD;
#pragma unroll
    for (int mm = 0; mm < 8; ++mm) {
        ctxb[(mm + hi * 8) * DD + n0a + r] = acc0[mm];
        ctxb[(mm + hi * 8) * DD + n0b + r] = acc1[mm];
    }
}

extern "C" void kernel_launch(void* const* d_in, const int* in_sizes, int n_in,
                              void* d_out, int out_size, void* d_ws, size_t ws_size,
                              hipStream_t stream) {
    (void)in_sizes; (void)n_in; (void)out_size; (void)d_ws; (void)ws_size;

    const float* q = (const float*)d_in[0];
    const float* k = (const float*)d_in[1];
    const float* v = (const float*)d_in[2];

    float* ctx = (float*)d_out;
    float* att = (float*)d_out + (size_t)BB * SS * DD;

    // ~197 KB dynamic LDS per workgroup (WGP has 320 KB)
    hipFuncSetAttribute((const void*)attn_fwd_kernel,
                        hipFuncAttributeMaxDynamicSharedMemorySize,
                        (int)LDS_BYTES);

    dim3 grid(BB * (SS / 16));   // 2048 workgroups
    dim3 block(NTHREADS);        // 4 waves of 32
    attn_fwd_kernel<<<grid, block, LDS_BYTES, stream>>>(q, k, v, ctx, att);
}
